// MSAOPM_20194936225883
// MI455X (gfx1250) — compile-verified, matrix operand order checked by reading, not run on hardware
//
#include <hip/hip_runtime.h>

// ---- sizes from the reference ----
#define NB   128   // N (batch, contraction dim of the outer product)
#define LSEQ 256   // L
#define DM   256   // M_DIM
#define CC   32    // C
#define ZD   128   // Z_DIM

typedef __bf16 bf16;
typedef bf16  v16bf __attribute__((ext_vector_type(16)));
typedef bf16  v8bf  __attribute__((ext_vector_type(8)));
typedef float v8f   __attribute__((ext_vector_type(8)));
typedef float v4f   __attribute__((ext_vector_type(4)));

// ------------------------------------------------------------------
// Phase A: layernorm + linear1/linear2, write bf16 transposed layouts
//   aT[l][c][n]  (scaled by 1/N, from w2 / "linear2" -> einsum 'a' side)
//   bT[l][c][n]  (from w1 / "linear1" -> einsum 'b' side)
// one wave per (n,l) row; 4 waves / block
// ------------------------------------------------------------------
__global__ __launch_bounds__(128) void msaopm_prep(
    const float* __restrict__ m, const float* __restrict__ ln_w,
    const float* __restrict__ ln_b,
    const float* __restrict__ w1, const float* __restrict__ b1,
    const float* __restrict__ w2, const float* __restrict__ b2,
    bf16* __restrict__ aT, bf16* __restrict__ bT)
{
    __shared__ float o_lds[4][DM];
    const int wid  = threadIdx.x >> 5;
    const int lane = threadIdx.x & 31;
    const int w    = blockIdx.x * 4 + wid;      // 0..32767 == n*256 + l
    const int n    = w >> 8;
    const int l    = w & 255;

    const float* row = m + (size_t)w * DM;
    v4f x0 = *(const v4f*)(row + lane * 8);
    v4f x1 = *(const v4f*)(row + lane * 8 + 4);

    float s = 0.f, sq = 0.f;
#pragma unroll
    for (int t = 0; t < 4; ++t) {
        s  += x0[t] + x1[t];
        sq += x0[t] * x0[t] + x1[t] * x1[t];
    }
#pragma unroll
    for (int off = 16; off >= 1; off >>= 1) {   // wave32 reduction
        s  += __shfl_xor(s,  off, 32);
        sq += __shfl_xor(sq, off, 32);
    }
    const float mean = s * (1.f / DM);
    const float var  = sq * (1.f / DM) - mean * mean;
    const float rstd = rsqrtf(var + 1e-5f);

#pragma unroll
    for (int t = 0; t < 4; ++t) {
        const int d0 = lane * 8 + t, d1 = lane * 8 + 4 + t;
        o_lds[wid][d0] = (x0[t] - mean) * rstd * ln_w[d0] + ln_b[d0];
        o_lds[wid][d1] = (x1[t] - mean) * rstd * ln_w[d1] + ln_b[d1];
    }
    __syncthreads();

    const int c = lane;            // one output channel per lane
    float oa = 0.f, ob = 0.f;
#pragma unroll 4
    for (int d = 0; d < DM; ++d) {
        const float o = o_lds[wid][d];
        oa = fmaf(o, w2[d * CC + c], oa);   // linear2 -> 'a'
        ob = fmaf(o, w1[d * CC + c], ob);   // linear1 -> 'b'
    }
    oa = (oa + b2[c]) * (1.f / NB);         // fold the 1/N (exact pow2 in bf16)
    ob =  ob + b1[c];

    aT[((size_t)l * CC + c) * NB + n] = (bf16)oa;
    bT[((size_t)l * CC + c) * NB + n] = (bf16)ob;
}

// ------------------------------------------------------------------
// w3 [1024,128] f32  ->  w3T [128][1024-permuted] bf16.
// The contraction index k = a*32+b is PERMUTED to kappa so that step-2's
// accumulator spill is contiguous per lane:
//   kappa = chunk*8 + r,  chunk = ((ma*2+nb)*2 + half)*16 + lml
//   with  M(a) = ma*16 + half*8 + r,  N(b) = nb*16 + lml
// Step 3 contracts pairbuf[p][kappa] * w3T[n][kappa]; a permutation applied
// to BOTH sides leaves the dot product invariant.
// ------------------------------------------------------------------
__global__ __launch_bounds__(256) void msaopm_w3t(
    const float* __restrict__ w3, bf16* __restrict__ w3T)
{
    const int t     = blockIdx.x * 256 + threadIdx.x;   // 0..131071
    const int n     = t & (ZD - 1);
    const int kappa = t >> 7;
    const int r     = kappa & 7;
    const int chunk = kappa >> 3;          // 0..127
    const int lml   = chunk & 15;
    const int half  = (chunk >> 4) & 1;
    const int nb    = (chunk >> 5) & 1;
    const int ma    = (chunk >> 6) & 1;
    const int Mi    = ma * 16 + half * 8 + r;   // 'a' index
    const int Ni    = nb * 16 + lml;            // 'b' index
    w3T[(size_t)n * (CC * CC) + kappa] =
        (bf16)w3[(size_t)(Mi * CC + Ni) * ZD + n];
}

// ------------------------------------------------------------------
// Fragment loader: 16-bit WMMA operand layout = two contiguous
// 8-element (16B) chunks per lane:  k = half*8 + e  and  k = 16 + half*8 + e
// caller passes base already offset by (row/col)*stride + kblock*32 + half*8
// ------------------------------------------------------------------
__device__ __forceinline__ v16bf load_frag(const bf16* base)
{
    v8bf lo = *(const v8bf*)(base);
    v8bf hi = *(const v8bf*)(base + 16);
    return __builtin_shufflevector(lo, hi,
        0, 1, 2, 3, 4, 5, 6, 7, 8, 9, 10, 11, 12, 13, 14, 15);
}

#define PAIR_STRIDE 1040   // 1024 + pad (bf16 elems); keeps 16B alignment

// ------------------------------------------------------------------
// Phase B (fused): per wave: i fixed, 16 j's.
//   step2: OP(i,j) 32x32 = A_i^T B_j (K=128) via 16x v_wmma_f32_16x16x32_bf16
//          -> bf16 into per-wave LDS rows [16][1024] (kappa-permuted,
//             4x ds_store_b128 per pair per lane)
//   step3: [16 x 1024] @ w3T -> z[i, j0..j0+15, 0..127]  (256 WMMAs)
// ------------------------------------------------------------------
__global__ __launch_bounds__(128) void msaopm_fused(
    const bf16* __restrict__ aT, const bf16* __restrict__ bT,
    const bf16* __restrict__ w3T, const float* __restrict__ b3,
    float* __restrict__ z)
{
    extern __shared__ bf16 smem[];              // 4 waves * 16 * PAIR_STRIDE
    const int wid  = threadIdx.x >> 5;
    const int lane = threadIdx.x & 31;
    const int half = lane >> 4;
    const int lml  = lane & 15;
    bf16* pairbuf = smem + (size_t)wid * 16 * PAIR_STRIDE;

    const int g  = blockIdx.x * 4 + wid;        // 0..4095
    const int i  = g >> 4;
    const int j0 = (g & 15) << 4;

    // A fragments for row-block i, reused across all 16 pairs
    v16bf afrag[2][4];
    const bf16* aBase = aT + (size_t)i * (CC * NB);
#pragma unroll
    for (int ma = 0; ma < 2; ++ma)
#pragma unroll
        for (int kb = 0; kb < 4; ++kb)
            afrag[ma][kb] =
                load_frag(aBase + (size_t)(ma * 16 + lml) * NB + kb * 32 + half * 8);

    // ---- step 2: 16 outer-product blocks ----
    for (int p = 0; p < 16; ++p) {
        const int j = j0 + p;
        const bf16* bBase =
            bT + (size_t)j * (CC * NB) + (size_t)lml * NB + half * 8;

        v8f acc[2][2] = {};
#pragma unroll
        for (int kb = 0; kb < 4; ++kb) {
#pragma unroll
            for (int nb = 0; nb < 2; ++nb) {
                v16bf bfrag = load_frag(bBase + (size_t)nb * 16 * NB + kb * 32);
#pragma unroll
                for (int ma = 0; ma < 2; ++ma)
                    acc[ma][nb] = __builtin_amdgcn_wmma_f32_16x16x32_bf16(
                        false, afrag[ma][kb], false, bfrag,
                        (short)0, acc[ma][nb], false, false);
            }
        }
        // spill OP block (already scaled by 1/N) to LDS as bf16 in the
        // kappa-permuted order: one contiguous 16B chunk per tile per lane
        bf16* prow = pairbuf + (size_t)p * PAIR_STRIDE;
#pragma unroll
        for (int ma = 0; ma < 2; ++ma)
#pragma unroll
            for (int nb = 0; nb < 2; ++nb) {
                v8bf packed;
#pragma unroll
                for (int r = 0; r < 8; ++r)
                    packed[r] = (bf16)acc[ma][nb][r];
                const int chunk = ((ma * 2 + nb) * 2 + half) * 16 + lml;
                *(v8bf*)(prow + chunk * 8) = packed;
            }
    }
    __syncthreads();

    // ---- step 3: [16 pairs x 1024] @ w3T[1024 x 128] (kappa space) ----
    v8f zacc[8] = {};
#pragma unroll 1
    for (int kb = 0; kb < 32; ++kb) {
        v16bf a2 = load_frag(pairbuf + (size_t)lml * PAIR_STRIDE + kb * 32 + half * 8);
#pragma unroll
        for (int ct = 0; ct < 8; ++ct) {
            v16bf wfrag = load_frag(
                w3T + (size_t)(ct * 16 + lml) * (CC * CC) + kb * 32 + half * 8);
            zacc[ct] = __builtin_amdgcn_wmma_f32_16x16x32_bf16(
                false, a2, false, wfrag, (short)0, zacc[ct], false, false);
        }
    }

    // ---- write z[i][j0+p][ncol] = acc + b3 ----
#pragma unroll
    for (int ct = 0; ct < 8; ++ct) {
        const int ncol = ct * 16 + lml;
        const float bias = b3[ncol];
#pragma unroll
        for (int r = 0; r < 8; ++r) {
            const int p = half * 8 + r;
            const int j = j0 + p;
            z[((size_t)i * LSEQ + j) * ZD + ncol] = zacc[ct][r] + bias;
        }
    }
}

// ------------------------------------------------------------------
extern "C" void kernel_launch(void* const* d_in, const int* in_sizes, int n_in,
                              void* d_out, int out_size, void* d_ws, size_t ws_size,
                              hipStream_t stream)
{
    (void)in_sizes; (void)n_in; (void)out_size; (void)ws_size;
    const float* m    = (const float*)d_in[0];
    const float* ln_w = (const float*)d_in[1];
    const float* ln_b = (const float*)d_in[2];
    const float* w1   = (const float*)d_in[3];
    const float* b1   = (const float*)d_in[4];
    const float* w2   = (const float*)d_in[5];
    const float* b2   = (const float*)d_in[6];
    const float* w3   = (const float*)d_in[7];
    const float* b3   = (const float*)d_in[8];
    float* z = (float*)d_out;

    char* ws   = (char*)d_ws;
    bf16* aT   = (bf16*)(ws);                  // 2 MB   [256][32][128]
    bf16* bT   = (bf16*)(ws + (2u << 20));     // 2 MB   [256][32][128]
    bf16* w3T  = (bf16*)(ws + (4u << 20));     // 256 KB [128][1024] (permuted)

    msaopm_prep<<<8192, 128, 0, stream>>>(m, ln_w, ln_b, w1, b1, w2, b2, aT, bT);
    msaopm_w3t<<<512, 256, 0, stream>>>(w3, w3T);

    const size_t smem = 4u * 16u * PAIR_STRIDE * sizeof(bf16); // 133,120 B / WGP
    msaopm_fused<<<1024, 128, smem, stream>>>(aT, bT, w3T, b3, z);
}
